// TorchBlock_29867202576549
// MI455X (gfx1250) — matrix operand order repaired and checked by reading, not compiled, not run on hardware
//
#include <hip/hip_runtime.h>
#include <hip/hip_bf16.h>

#define DIM 1024
#define HEADS 16
#define HD 64
#define SEQ 4096
#define FFN 4096

typedef __attribute__((ext_vector_type(16))) __bf16 v16bf;
typedef __attribute__((ext_vector_type(8))) float v8f;

union FragBF {
  uint4 q[2];
  v16bf v;
};

__device__ __forceinline__ unsigned short f2bf(float f) {
  unsigned int u = __float_as_uint(f);
  u += 0x7FFFu + ((u >> 16) & 1u);
  return (unsigned short)(u >> 16);
}

__device__ __forceinline__ v8f zero8() {
  v8f z;
#pragma unroll
  for (int i = 0; i < 8; ++i) z[i] = 0.0f;
  return z;
}

// A fragment (16x32 bf16) from row-major source, leading dim ld (elements).
// Per ISA: lanes 0-15 hold row M=lane, K chunks {0..7, 16..23};
//          lanes 16-31 hold row M=lane-16, K chunks {8..15, 24..31}.
__device__ __forceinline__ v16bf load_frag_a(const unsigned short* base, int ld, int lane) {
  const int row = lane & 15;
  const int half = lane >> 4;
  FragBF f;
  f.q[0] = *(const uint4*)(base + (size_t)row * ld + half * 8);
  f.q[1] = *(const uint4*)(base + (size_t)row * ld + 16 + half * 8);
  return f.v;
}

// B fragment (32x16 bf16) from N-major (transposed) source: baseT[n*ld + k].
// lanes 0-15: col N=lane, K chunks {0..7, 16..23}; lanes 16-31: col N=lane-16, K chunks {8..15, 24..31}.
__device__ __forceinline__ v16bf load_frag_b(const unsigned short* baseT, int ld, int lane) {
  const int col = lane & 15;
  const int half = lane >> 4;
  FragBF f;
  f.q[0] = *(const uint4*)(baseT + (size_t)col * ld + half * 8);
  f.q[1] = *(const uint4*)(baseT + (size_t)col * ld + 16 + half * 8);
  return f.v;
}

__device__ __forceinline__ v8f wmma_bf16(v16bf a, v16bf b, v8f c) {
  return __builtin_amdgcn_wmma_f32_16x16x32_bf16(false, a, false, b, (short)0, c, false, false);
}

// Async global->LDS copy of 64 contiguous bytes (4 x B128); INST_OFFSET is
// applied to both the LDS destination and the global source, so one LDS
// address VGPR + one global address VGPR pair serve all four transfers.
__device__ __forceinline__ void async_copy64(const unsigned short* g, unsigned short* l) {
  unsigned lo = (unsigned)(uintptr_t)l;  // low 32 bits of generic ptr == LDS byte address
  asm volatile(
      "global_load_async_to_lds_b128 %0, %1, off\n\t"
      "global_load_async_to_lds_b128 %0, %1, off offset:16\n\t"
      "global_load_async_to_lds_b128 %0, %1, off offset:32\n\t"
      "global_load_async_to_lds_b128 %0, %1, off offset:48"
      :
      : "v"(lo), "v"(g)
      : "memory");
}

__device__ __forceinline__ void wait_async0() {
  asm volatile("s_wait_asynccnt 0" ::: "memory");
}

// ---------------- conversion kernels ----------------

__global__ void cvt_f32_bf16(const float* __restrict__ in, unsigned short* __restrict__ out, int n) {
  int i = blockIdx.x * blockDim.x + threadIdx.x;
  if (i < n) out[i] = f2bf(in[i]);
}

// W [K,N] fp32 row-major -> Wt [N,K] bf16 row-major
__global__ void cvt_transpose_bf16(const float* __restrict__ W, unsigned short* __restrict__ Wt,
                                   int K, int N) {
  int i = blockIdx.x * blockDim.x + threadIdx.x;
  if (i < K * N) {
    int n = i / K;
    int k = i - n * K;
    Wt[i] = f2bf(W[(size_t)k * N + n]);
  }
}

// ---------------- generic WMMA GEMM (async double-buffered) ----------------

enum { EPI_BF16 = 0, EPI_BF16T, EPI_TANH_RES, EPI_RELU, EPI_TANH_FINAL };

// C[M,N] = A[M,K](bf16,row) * Bt[N,K](bf16,row-as-N-major). Block tile 128x128,
// 8 waves of wave32, each wave computes 64x32. K stepped by 64 through
// double-buffered LDS filled with GLOBAL_LOAD_ASYNC_TO_LDS_B128.
template <int EPI>
__global__ __launch_bounds__(256) void gemm_bf16(
    const unsigned short* __restrict__ A, int lda,
    const unsigned short* __restrict__ Bt, int ldb,
    int K,
    unsigned short* __restrict__ outB, int ldob,
    float* __restrict__ outF, int ldof,
    const float* __restrict__ resid, int ldr) {
  __shared__ unsigned short As[2][128 * 72];  // 72-elem stride: 144B rows, 16B aligned
  __shared__ unsigned short Bs[2][128 * 72];

  const int tid = threadIdx.x;
  const int lane = tid & 31;
  const int wave = tid >> 5;
  const int wr = wave >> 2;  // 0..1 -> 64-row slab
  const int wc = wave & 3;   // 0..3 -> 32-col slab
  const int rowBase = blockIdx.y * 128;
  const int colBase = blockIdx.x * 128;

  v8f acc[4][2];
#pragma unroll
  for (int i = 0; i < 4; ++i)
#pragma unroll
    for (int j = 0; j < 2; ++j) acc[i][j] = zero8();

  // staging: 128 rows x 64 elems (128B/row); 2 threads per row, 64B each
  const int stRow = tid >> 1;
  const int stCol = (tid & 1) * 32;  // element offset within the 64-wide slab

  auto stage = [&](int k0, int buf) {
    const unsigned short* gA = A + (size_t)(rowBase + stRow) * lda + k0 + stCol;
    const unsigned short* gB = Bt + (size_t)(colBase + stRow) * ldb + k0 + stCol;
    async_copy64(gA, &As[buf][stRow * 72 + stCol]);
    async_copy64(gB, &Bs[buf][stRow * 72 + stCol]);
  };

  stage(0, 0);
  wait_async0();
  __syncthreads();

  const int nSteps = K >> 6;
  for (int s = 0; s < nSteps; ++s) {
    const int cur = s & 1;
    if (s + 1 < nSteps) stage((s + 1) << 6, cur ^ 1);

#pragma unroll
    for (int kk = 0; kk < 2; ++kk) {
      v16bf a[4];
#pragma unroll
      for (int rs = 0; rs < 4; ++rs)
        a[rs] = load_frag_a(&As[cur][(wr * 64 + rs * 16) * 72 + kk * 32], 72, lane);
      v16bf b[2];
#pragma unroll
      for (int cs = 0; cs < 2; ++cs)
        b[cs] = load_frag_b(&Bs[cur][(wc * 32 + cs * 16) * 72 + kk * 32], 72, lane);
#pragma unroll
      for (int rs = 0; rs < 4; ++rs)
#pragma unroll
        for (int cs = 0; cs < 2; ++cs) acc[rs][cs] = wmma_bf16(a[rs], b[cs], acc[rs][cs]);
    }

    wait_async0();
    __syncthreads();
  }

  const int half = lane >> 4;
  const int l15 = lane & 15;
#pragma unroll
  for (int rs = 0; rs < 4; ++rs) {
#pragma unroll
    for (int cs = 0; cs < 2; ++cs) {
#pragma unroll
      for (int r = 0; r < 8; ++r) {
        int grow = rowBase + wr * 64 + rs * 16 + r + half * 8;
        int gcol = colBase + wc * 32 + cs * 16 + l15;
        float v = acc[rs][cs][r];
        if (EPI == EPI_BF16) {
          outB[(size_t)grow * ldob + gcol] = f2bf(v);
        } else if (EPI == EPI_BF16T) {
          outB[(size_t)gcol * ldob + grow] = f2bf(v);
        } else if (EPI == EPI_TANH_RES) {
          float y = tanhf(resid[(size_t)grow * ldr + gcol] + v);
          outF[(size_t)grow * ldof + gcol] = y;
          outB[(size_t)grow * ldob + gcol] = f2bf(y);
        } else if (EPI == EPI_RELU) {
          outB[(size_t)grow * ldob + gcol] = f2bf(v > 0.0f ? v : 0.0f);
        } else {  // EPI_TANH_FINAL
          outF[(size_t)grow * ldof + gcol] = tanhf(resid[(size_t)grow * ldr + gcol] + v);
        }
      }
    }
  }
}

// ---------------- flash attention (causal, online softmax) ----------------
// Q,K: bf16 [SEQ, DIM]; Vt: bf16 [DIM, SEQ] (head-major rows h*64+hd);
// O: bf16 [SEQ, DIM]. One wave handles 16 query rows of one head.
__global__ __launch_bounds__(128) void attn_kernel(
    const unsigned short* __restrict__ Q,
    const unsigned short* __restrict__ Kc,
    const unsigned short* __restrict__ Vt,
    unsigned short* __restrict__ O) {
  __shared__ unsigned short Plds[4][16 * 72];  // wave-private 16x64 P tiles, padded

  const int tid = threadIdx.x;
  const int lane = tid & 31;
  const int wave = tid >> 5;
  const int h = blockIdx.y;
  const int qr = blockIdx.x * 64 + wave * 16;
  const int half = lane >> 4;
  const int l15 = lane & 15;

  const unsigned short* qbase = Q + (size_t)qr * DIM + h * HD;
  v16bf aq0 = load_frag_a(qbase, DIM, lane);       // K dim 0..31 of head
  v16bf aq1 = load_frag_a(qbase + 32, DIM, lane);  // K dim 32..63

  v8f o[4];
#pragma unroll
  for (int nt = 0; nt < 4; ++nt) o[nt] = zero8();
  float m[8], l[8];
#pragma unroll
  for (int r = 0; r < 8; ++r) {
    m[r] = -3.0e38f;
    l[r] = 0.0f;
  }

  unsigned short* P = Plds[wave];
  const int ktMax = (qr + 15) >> 6;  // causal limit

  for (int kt = 0; kt <= ktMax; ++kt) {
    const int kvb = kt * 64;
    // ---- S = Q K^T for a 16x64 tile ----
    v8f s[4];
#pragma unroll
    for (int j = 0; j < 4; ++j) {
      const unsigned short* kb = Kc + (size_t)(kvb + j * 16) * DIM + h * HD;
      v16bf b0 = load_frag_b(kb, DIM, lane);
      v16bf b1 = load_frag_b(kb + 32, DIM, lane);
      v8f acc = zero8();
      acc = wmma_bf16(aq0, b0, acc);
      acc = wmma_bf16(aq1, b1, acc);
      s[j] = acc;
    }

    const bool domask = (kvb + 63) > qr;
    const float scale = 0.125f;  // 1/sqrt(64)

#pragma unroll
    for (int r = 0; r < 8; ++r) {
      const int grow = qr + r + half * 8;  // global query row for this C slot
      float v[4];
      float mx = -3.0e38f;
#pragma unroll
      for (int j = 0; j < 4; ++j) {
        float x = s[j][r] * scale;
        if (domask && (kvb + j * 16 + l15) > grow) x = -3.0e38f;
        v[j] = x;
        mx = fmaxf(mx, x);
      }
      for (int off = 1; off < 16; off <<= 1) mx = fmaxf(mx, __shfl_xor(mx, off, 32));
      float newm = fmaxf(m[r], mx);
      float psum = 0.0f;
#pragma unroll
      for (int j = 0; j < 4; ++j) {
        float p = __expf(v[j] - newm);
        psum += p;
        P[(r + half * 8) * 72 + j * 16 + l15] = f2bf(p);
      }
      for (int off = 1; off < 16; off <<= 1) psum += __shfl_xor(psum, off, 32);
      float corr = __expf(m[r] - newm);
      l[r] = l[r] * corr + psum;
      m[r] = newm;
#pragma unroll
      for (int nt = 0; nt < 4; ++nt) o[nt][r] *= corr;
    }

    __builtin_amdgcn_wave_barrier();
    asm volatile("s_wait_dscnt 0" ::: "memory");

    // ---- O += P (16x64) x V (64x64), P reloaded in A-fragment layout ----
    v16bf ap0 = load_frag_a(P, 72, lane);
    v16bf ap1 = load_frag_a(P + 32, 72, lane);
#pragma unroll
    for (int nt = 0; nt < 4; ++nt) {
      const unsigned short* vb = Vt + (size_t)(h * HD + nt * 16) * SEQ + kvb;
      v16bf bv0 = load_frag_b(vb, SEQ, lane);
      v16bf bv1 = load_frag_b(vb + 32, SEQ, lane);
      o[nt] = wmma_bf16(ap0, bv0, o[nt]);
      o[nt] = wmma_bf16(ap1, bv1, o[nt]);
    }
    __builtin_amdgcn_wave_barrier();
  }

#pragma unroll
  for (int r = 0; r < 8; ++r) {
    const float inv = 1.0f / l[r];
    const int grow = qr + r + half * 8;
#pragma unroll
    for (int nt = 0; nt < 4; ++nt) {
      O[(size_t)grow * DIM + h * HD + nt * 16 + l15] = f2bf(o[nt][r] * inv);
    }
  }
}

// ---------------- host orchestration ----------------

extern "C" void kernel_launch(void* const* d_in, const int* in_sizes, int n_in,
                              void* d_out, int out_size, void* d_ws, size_t ws_size,
                              hipStream_t stream) {
  const float* x = (const float*)d_in[0];
  const float* Wq = (const float*)d_in[1];
  const float* Wk = (const float*)d_in[2];
  const float* Wv = (const float*)d_in[3];
  const float* Wo = (const float*)d_in[4];
  const float* W1 = (const float*)d_in[5];
  const float* W2 = (const float*)d_in[6];
  float* out = (float*)d_out;

  unsigned char* ws = (unsigned char*)d_ws;
  size_t off = 0;
  auto alloc = [&](size_t bytes) -> void* {
    void* p = ws + off;
    off += (bytes + 255) & ~(size_t)255;
    return p;
  };
  unsigned short* xb = (unsigned short*)alloc((size_t)SEQ * DIM * 2);
  unsigned short* wqT = (unsigned short*)alloc((size_t)DIM * DIM * 2);
  unsigned short* wkT = (unsigned short*)alloc((size_t)DIM * DIM * 2);
  unsigned short* wvT = (unsigned short*)alloc((size_t)DIM * DIM * 2);
  unsigned short* woT = (unsigned short*)alloc((size_t)DIM * DIM * 2);
  unsigned short* w1T = (unsigned short*)alloc((size_t)DIM * FFN * 2);
  unsigned short* w2T = (unsigned short*)alloc((size_t)FFN * DIM * 2);
  unsigned short* Qb = (unsigned short*)alloc((size_t)SEQ * DIM * 2);
  unsigned short* Kb = (unsigned short*)alloc((size_t)SEQ * DIM * 2);
  unsigned short* VtB = (unsigned short*)alloc((size_t)DIM * SEQ * 2);
  unsigned short* attnB = (unsigned short*)alloc((size_t)SEQ * DIM * 2);
  float* yF = (float*)alloc((size_t)SEQ * DIM * 4);
  unsigned short* yB = (unsigned short*)alloc((size_t)SEQ * DIM * 2);
  unsigned short* hB = (unsigned short*)alloc((size_t)SEQ * FFN * 2);

  // precision downconvert + weight transposes
  cvt_f32_bf16<<<(SEQ * DIM + 255) / 256, 256, 0, stream>>>(x, xb, SEQ * DIM);
  cvt_transpose_bf16<<<(DIM * DIM + 255) / 256, 256, 0, stream>>>(Wq, wqT, DIM, DIM);
  cvt_transpose_bf16<<<(DIM * DIM + 255) / 256, 256, 0, stream>>>(Wk, wkT, DIM, DIM);
  cvt_transpose_bf16<<<(DIM * DIM + 255) / 256, 256, 0, stream>>>(Wv, wvT, DIM, DIM);
  cvt_transpose_bf16<<<(DIM * DIM + 255) / 256, 256, 0, stream>>>(Wo, woT, DIM, DIM);
  cvt_transpose_bf16<<<(DIM * FFN + 255) / 256, 256, 0, stream>>>(W1, w1T, DIM, FFN);
  cvt_transpose_bf16<<<(FFN * DIM + 255) / 256, 256, 0, stream>>>(W2, w2T, FFN, DIM);

  dim3 gProj(DIM / 128, SEQ / 128);
  // Q = x Wq, K = x Wk (bf16 row-major); V stored transposed for the PV GEMM
  gemm_bf16<EPI_BF16><<<gProj, 256, 0, stream>>>(xb, DIM, wqT, DIM, DIM, Qb, DIM,
                                                 nullptr, 0, nullptr, 0);
  gemm_bf16<EPI_BF16><<<gProj, 256, 0, stream>>>(xb, DIM, wkT, DIM, DIM, Kb, DIM,
                                                 nullptr, 0, nullptr, 0);
  gemm_bf16<EPI_BF16T><<<gProj, 256, 0, stream>>>(xb, DIM, wvT, DIM, DIM, VtB, SEQ,
                                                  nullptr, 0, nullptr, 0);

  attn_kernel<<<dim3(SEQ / 64, HEADS), 128, 0, stream>>>(Qb, Kb, VtB, attnB);

  // y = tanh(x + attn @ Wo); keep f32 (final residual) and bf16 (FFN input)
  gemm_bf16<EPI_TANH_RES><<<gProj, 256, 0, stream>>>(attnB, DIM, woT, DIM, DIM, yB, DIM,
                                                     yF, DIM, x, DIM);
  // h = relu(y @ W1)
  gemm_bf16<EPI_RELU><<<dim3(FFN / 128, SEQ / 128), 256, 0, stream>>>(
      yB, DIM, w1T, DIM, DIM, hB, FFN, nullptr, 0, nullptr, 0);
  // out = tanh(y + h @ W2)
  gemm_bf16<EPI_TANH_FINAL><<<dim3(DIM / 128, SEQ / 128), 256, 0, stream>>>(
      hB, FFN, w2T, FFN, FFN, nullptr, 0, out, DIM, yF, DIM);
}
